// ClusterMerging_26929444946666
// MI455X (gfx1250) — matrix operand order repaired and checked by reading, not compiled, not run on hardware
//
#include <hip/hip_runtime.h>
#include <hip/hip_bf16.h>

typedef __attribute__((ext_vector_type(16))) __bf16 v16bf;
typedef __attribute__((ext_vector_type(8)))  float  v8f;

#define B_  8
#define N_  16384
#define C_  256
#define D_  2
#define K_  4096
#define F_  258          // C + D
#define O_  512          // 2C
#define EPS_ 1e-5f

// ---------------- workspace layout (bytes) ----------------
#define WS_POSMAX   0                         // 2 x u32
#define WS_COUNT    256                       // B*K floats       (131072 B)
#define WS_POSSUM   (256 + 131072)            // B*K*2 floats     (262144 B)
#define WS_INVSCALE (WS_POSSUM + 262144)      // B*K floats       (131072 B)
#define WS_FRAGB    (WS_INVSCALE + 131072)    // 8*2*512*8 dwords (262144 B)
#define WS_WEXT     (WS_FRAGB + 262144)       // 2*512 floats     (4096 B)
#define WS_ZERO_FLOATS ((WS_INVSCALE) / 4)    // zero posmax+count+possum

// ---------------- generic zero ----------------
__global__ void k_zero(float* __restrict__ p, long n) {
    long i = (long)blockIdx.x * blockDim.x + threadIdx.x;
    long stride = (long)gridDim.x * blockDim.x;
    for (; i < n; i += stride) p[i] = 0.0f;
}

// ---------------- global per-dim max of pos ----------------
__global__ void k_posmax(const float* __restrict__ pos, unsigned int* __restrict__ pmax) {
    int i = blockIdx.x * blockDim.x + threadIdx.x;
    int stride = gridDim.x * blockDim.x;
    float m0 = 0.0f, m1 = 0.0f;
    for (int p = i; p < B_ * N_; p += stride) {
        m0 = fmaxf(m0, pos[2 * p]);
        m1 = fmaxf(m1, pos[2 * p + 1]);
    }
    #pragma unroll
    for (int off = 16; off; off >>= 1) {
        m0 = fmaxf(m0, __shfl_xor(m0, off, 32));
        m1 = fmaxf(m1, __shfl_xor(m1, off, 32));
    }
    if ((threadIdx.x & 31) == 0) {            // positive floats: bit pattern is monotone
        atomicMax(&pmax[0], __float_as_uint(m0));
        atomicMax(&pmax[1], __float_as_uint(m1));
    }
}

// ---------------- per-cluster count + normalized-pos sums ----------------
__global__ void k_count(const float* __restrict__ pos, const int* __restrict__ asn,
                        const unsigned int* __restrict__ pmax,
                        float* __restrict__ count, float* __restrict__ possum) {
    int p = blockIdx.x * blockDim.x + threadIdx.x;
    if (p >= B_ * N_) return;
    int b = p / N_;
    float inv0 = 1.0f / __uint_as_float(pmax[0]);
    float inv1 = 1.0f / __uint_as_float(pmax[1]);
    int seg = b * K_ + asn[p];
    atomicAdd(&count[seg], 1.0f);
    atomicAdd(&possum[2 * seg],     pos[2 * p]     * inv0);
    atomicAdd(&possum[2 * seg + 1], pos[2 * p + 1] * inv1);
}

// ---------------- cluster mean pos, validity, inverse scale ----------------
__global__ void k_mean(const float* __restrict__ count, const float* __restrict__ possum,
                       float* __restrict__ out_pos, float* __restrict__ out_mask,
                       float* __restrict__ invscale) {
    int s = blockIdx.x * blockDim.x + threadIdx.x;
    if (s >= B_ * K_) return;
    float c = count[s];
    float safe = fmaxf(c, 1.0f);
    float valid = (c > 0.0f) ? 1.0f : 0.0f;
    out_pos[2 * s]     = possum[2 * s]     / safe;   // == mean*valid (empty => 0)
    out_pos[2 * s + 1] = possum[2 * s + 1] / safe;
    out_mask[s] = valid;
    invscale[s] = valid / safe;
}

// ---------------- repack lin_w into bf16 WMMA B-fragments ----------------
// fragB dword index i = ((kc*2 + khalf)*512 + o)*8 + vj
// dword holds bf16 pair (w[o][kbase+2vj], w[o][kbase+2vj+1]), kbase = kc*32 + khalf*16
__global__ void k_repack(const float* __restrict__ lin_w,
                         unsigned int* __restrict__ fragB, float* __restrict__ wext) {
    int i = blockIdx.x * blockDim.x + threadIdx.x;
    if (i < 8 * 2 * O_ * 8) {
        int vj = i & 7;
        int o  = (i >> 3) & (O_ - 1);
        int kh = (i >> 12) & 1;
        int kc = i >> 13;
        int kb = kc * 32 + kh * 16 + 2 * vj;
        union { __bf16 h[2]; unsigned int u; } cv;
        cv.h[0] = (__bf16)lin_w[o * F_ + kb];
        cv.h[1] = (__bf16)lin_w[o * F_ + kb + 1];
        fragB[i] = cv.u;
    } else {
        int j = i - 8 * 2 * O_ * 8;
        if (j < 2 * O_) {
            int d = j >> 9, o = j & (O_ - 1);
            wext[j] = lin_w[o * F_ + C_ + d];
        }
    }
}

// ---------------- fused LN + WMMA GEMM + cluster scatter-add ----------------
__global__ __launch_bounds__(256) void k_main(
        const float* __restrict__ feat, const float* __restrict__ pos,
        const int* __restrict__ asn, const float* __restrict__ norm_w,
        const float* __restrict__ norm_b, const unsigned int* __restrict__ pmax,
        const float* __restrict__ mean_pos, const unsigned int* __restrict__ fragB,
        const float* __restrict__ wext, float* __restrict__ out_feat) {
    // A-fragments for 16 points x 256 K, bf16, laid out per WMMA A lane mapping:
    // half index = (((kc*2 + grp)*16 + m)*8 + vj)*2 + lo
    __shared__ __align__(32) __bf16 lds_a[8 * 2 * 16 * 8 * 2];   // 8 KB
    __shared__ float lds_rel[2][16];
    __shared__ int   lds_seg[16];

    const int tid   = threadIdx.x;
    const int batch = blockIdx.x >> 10;          // N_/16 = 1024 point-tiles per batch
    const int pbase = (blockIdx.x & 1023) * 16;

    // ---- Phase 1: LayerNorm of 16 points (16 threads per point) ----
    const int p   = tid >> 4;
    const int l16 = tid & 15;
    const int pt  = batch * N_ + pbase + p;
    const int seg = batch * K_ + asn[pt];
    const float inv0 = 1.0f / __uint_as_float(pmax[0]);
    const float inv1 = 1.0f / __uint_as_float(pmax[1]);
    const float rx = pos[2 * pt]     * inv0 - mean_pos[2 * seg];
    const float ry = pos[2 * pt + 1] * inv1 - mean_pos[2 * seg + 1];
    if (l16 == 0) lds_seg[p] = seg;

    float x[17];
    float s = 0.0f, s2 = 0.0f;
    #pragma unroll
    for (int i = 0; i < 17; ++i) {
        int j = l16 + 16 * i;
        float v = 0.0f;
        if (j < F_) v = (j < C_) ? feat[(long)pt * C_ + j] : ((j == C_) ? rx : ry);
        x[i] = v;
        s += v; s2 += v * v;
    }
    #pragma unroll
    for (int off = 8; off; off >>= 1) {
        s  += __shfl_xor(s,  off, 16);
        s2 += __shfl_xor(s2, off, 16);
    }
    const float mu   = s  * (1.0f / F_);
    const float var  = s2 * (1.0f / F_) - mu * mu;
    const float rsig = rsqrtf(var + EPS_);
    #pragma unroll
    for (int i = 0; i < 17; ++i) {
        int j = l16 + 16 * i;
        if (j >= F_) continue;
        float xn = (x[i] - mu) * rsig * norm_w[j] + norm_b[j];
        if (j < C_) {
            int kc = j >> 5, kk = j & 31;
            int grp = kk >> 4;
            int vj  = (kk & 15) >> 1;
            int lo  = kk & 1;
            lds_a[((((kc * 2 + grp) * 16 + p) * 8) + vj) * 2 + lo] = (__bf16)xn;
        } else {
            lds_rel[j - C_][p] = xn;
        }
    }
    __syncthreads();

    // ---- Phase 2: each wave owns 64 output columns = 4 WMMA N-tiles ----
    const int w   = tid >> 5;
    const int l   = tid & 31;
    const int m   = l & 15;      // A row / B column within tile
    const int grp = l >> 4;      // A K-group / B K-half / C row-group
    const int obase = w * 64 + m;

    v8f acc[4] = {v8f{}, v8f{}, v8f{}, v8f{}};
    #pragma unroll
    for (int kc = 0; kc < 8; ++kc) {
        const v16bf af = *(const v16bf*)&lds_a[((kc * 2 + grp) * 16 + m) * 16];
        const unsigned int* bb = fragB + (long)((kc * 2 + grp) * O_) * 8;
        #pragma unroll
        for (int t = 0; t < 4; ++t) {
            const v16bf bf = *(const v16bf*)(bb + (obase + t * 16) * 8);
            acc[t] = __builtin_amdgcn_wmma_f32_16x16x32_bf16(
                false, af, false, bf, (short)0, acc[t], false, false);
        }
    }

    // rank-2 update for the 2 rel_pos columns + scatter-add into cluster sums
    #pragma unroll
    for (int t = 0; t < 4; ++t) {
        const int o = obase + t * 16;
        const float w0 = wext[o], w1 = wext[O_ + o];
        #pragma unroll
        for (int r = 0; r < 8; ++r) {
            const int mr = grp * 8 + r;
            const float v = acc[t][r] + lds_rel[0][mr] * w0 + lds_rel[1][mr] * w1;
            atomicAdd(&out_feat[(long)lds_seg[mr] * O_ + o], v);
        }
    }
}

// ---------------- final mean-pool scaling (valid/count), in place ----------------
__global__ void k_scale(float* __restrict__ out_feat, const float* __restrict__ invscale) {
    long i = (long)blockIdx.x * blockDim.x + threadIdx.x;
    long stride = (long)gridDim.x * blockDim.x;
    long n = (long)B_ * K_ * O_;
    for (; i < n; i += stride) out_feat[i] *= invscale[i >> 9];
}

extern "C" void kernel_launch(void* const* d_in, const int* in_sizes, int n_in,
                              void* d_out, int out_size, void* d_ws, size_t ws_size,
                              hipStream_t stream) {
    (void)in_sizes; (void)n_in; (void)out_size; (void)ws_size;
    const float* pos    = (const float*)d_in[0];
    const float* feat   = (const float*)d_in[1];
    const int*   asn    = (const int*)  d_in[2];
    const float* norm_w = (const float*)d_in[3];
    const float* norm_b = (const float*)d_in[4];
    const float* lin_w  = (const float*)d_in[5];

    float* out_pos  = (float*)d_out;                        // B*K*2
    float* out_feat = out_pos + (long)B_ * K_ * 2;          // B*K*512
    float* out_mask = out_feat + (long)B_ * K_ * O_;        // B*K

    char* ws = (char*)d_ws;
    unsigned int* pmax     = (unsigned int*)(ws + WS_POSMAX);
    float*        count    = (float*)(ws + WS_COUNT);
    float*        possum   = (float*)(ws + WS_POSSUM);
    float*        invscale = (float*)(ws + WS_INVSCALE);
    unsigned int* fragB    = (unsigned int*)(ws + WS_FRAGB);
    float*        wext     = (float*)(ws + WS_WEXT);

    // 1) zero ws head (posmax/count/possum) and out_feat accumulator
    k_zero<<<512, 256, 0, stream>>>((float*)ws, (long)WS_ZERO_FLOATS);
    k_zero<<<2048, 256, 0, stream>>>(out_feat, (long)B_ * K_ * O_);
    // 2) global position max
    k_posmax<<<256, 256, 0, stream>>>(pos, pmax);
    // 3) cluster counts + normalized position sums
    k_count<<<(B_ * N_) / 256, 256, 0, stream>>>(pos, asn, pmax, count, possum);
    // 4) cluster means / mask / pooling scale (mean written to out_pos, reused as gather src)
    k_mean<<<(B_ * K_) / 256, 256, 0, stream>>>(count, possum, out_pos, out_mask, invscale);
    // 5) repack lin_w into bf16 WMMA B-fragments + extra columns
    k_repack<<<(8 * 2 * O_ * 8 + 2 * O_ + 255) / 256, 256, 0, stream>>>(lin_w, fragB, wext);
    // 6) fused LN + WMMA GEMM + scatter-add (8192 point-tiles of 16)
    k_main<<<B_ * (N_ / 16), 256, 0, stream>>>(feat, pos, asn, norm_w, norm_b, pmax,
                                               out_pos, fragB, wext, out_feat);
    // 7) divide pooled sums by count, mask invalid clusters
    k_scale<<<2048, 256, 0, stream>>>(out_feat, invscale);
}
